// HyperbolicRationalFourierOperator_8358006358663
// MI455X (gfx1250) — matrix operand order repaired and checked by reading, not compiled, not run on hardware
//
#include <hip/hip_runtime.h>

typedef __attribute__((ext_vector_type(16))) _Float16 v16h;
typedef __attribute__((ext_vector_type(8)))  _Float16 v8h;
typedef __attribute__((ext_vector_type(4)))  _Float16 v4h;
typedef __attribute__((ext_vector_type(8)))  float    v8f;
typedef __attribute__((ext_vector_type(4)))  float    v4f;

#define DIM   512
#define NHEAD 8
#define HD    64
#define NTOK  2048
#define BSZ   8            // b*s
#define MROWS (BSZ*NTOK)   // 16384
#define QBLK  128          // queries per attention block (4 waves x 32)

// ---------------------------------------------------------------------------
// WMMA helpers (CDNA5 wave32, 16x16x32 f16 -> f32)
// ---------------------------------------------------------------------------
__device__ __forceinline__ v8f wmma_f16(v16h a, v16h b, v8f c) {
  return __builtin_amdgcn_wmma_f32_16x16x32_f16(false, a, false, b, (short)0, c,
                                                false, false);
}

__device__ __forceinline__ v8f zero8() {
  v8f z;
#pragma unroll
  for (int i = 0; i < 8; ++i) z[i] = 0.0f;
  return z;
}

// A fragment 16x32 (M x K), source row-major with leading dim lda.
// lane = row (lane&15); lanes 0-15 hold K = 0..15, lanes 16-31 hold
// K = 16..31 (two contiguous 16B loads per lane). Works for global
// or LDS pointers (LDS -> ds_load_b128).
__device__ __forceinline__ v16h load_A16x32(const _Float16* base, int lda) {
  const int l = threadIdx.x & 31;
  const _Float16* p = base + (size_t)(l & 15) * lda + ((l >> 4) << 4);
  v8h lo = *(const v8h*)p;
  v8h hi = *(const v8h*)(p + 8);
  return __builtin_shufflevector(lo, hi, 0, 1, 2, 3, 4, 5, 6, 7,
                                 8, 9, 10, 11, 12, 13, 14, 15);
}

// B fragment 32x16 (K x N) where COLUMN n is contiguous in memory, i.e. the
// source is stored row-major as [N][K] with leading dim ldt.
// lane = column (lane&15); lanes 0-15 hold K=0..15, lanes 16-31 hold K=16..31.
__device__ __forceinline__ v16h load_B32x16(const _Float16* baseT, int ldt) {
  const int l = threadIdx.x & 31;
  const _Float16* p = baseT + (size_t)(l & 15) * ldt + ((l >> 4) << 4);
  v8h lo = *(const v8h*)p;
  v8h hi = *(const v8h*)(p + 8);
  return __builtin_shufflevector(lo, hi, 0, 1, 2, 3, 4, 5, 6, 7,
                                 8, 9, 10, 11, 12, 13, 14, 15);
}

// lane <-> lane^16 exchange (wave32) via LDS crossbar, no memory touched.
__device__ __forceinline__ float xor16f(float x) {
  const int l = threadIdx.x & 31;
  int r = __builtin_amdgcn_ds_bpermute(((l ^ 16) << 2),
                                       __builtin_bit_cast(int, x));
  return __builtin_bit_cast(float, r);
}

// Async global -> LDS copy, 16 bytes per lane (CDNA5 VGLOBAL async encoding:
// VDST = LDS byte address, VADDR = 64-bit global address). Tracked by ASYNCcnt.
__device__ __forceinline__ void async_b128(unsigned int lds_off, const void* g) {
  asm volatile("global_load_async_to_lds_b128 %0, %1, off"
               :: "v"(lds_off), "v"((unsigned long long)(uintptr_t)g)
               : "memory");
}

// ---------------------------------------------------------------------------
// f32 -> f16 conversion (x and weights), 4 elements per thread
// ---------------------------------------------------------------------------
__global__ void cvt_f32_f16(const float* __restrict__ s,
                            _Float16* __restrict__ d, int n4) {
  int i = blockIdx.x * blockDim.x + threadIdx.x;
  if (i < n4) {
    v4f f = *(const v4f*)(s + (size_t)i * 4);
    v4h h;
#pragma unroll
    for (int j = 0; j < 4; ++j) h[j] = (_Float16)f[j];
    *(v4h*)(d + (size_t)i * 4) = h;
  }
}

// ---------------------------------------------------------------------------
// Q/K projection, computed TRANSPOSED:  D[o, m] = sum_k W[o,k] * x[m,k] + b[o]
// A = W (row-major, lane=channel row), B = x^T (columns = x rows, contiguous).
// D layout: lane = token, rows = channel -> row-major [token, hd] store is a
// single contiguous v8h per tile. Output: out[(bs*8+head)*2048 + tok][hd].
// Wave tile: 64 channels x 32 tokens. grid = (16384/32, 512/64)
// ---------------------------------------------------------------------------
__global__ void __launch_bounds__(32)
qk_proj_kernel(const _Float16* __restrict__ xh, const _Float16* __restrict__ W,
               const float* __restrict__ bias, _Float16* __restrict__ out) {
  const int l  = threadIdx.x & 31;
  const int t0 = blockIdx.x * 32;   // token group
  const int c0 = blockIdx.y * 64;   // channel group

  v8f acc[4][2];
#pragma unroll
  for (int mt = 0; mt < 4; ++mt) {
    const float* bp = bias + c0 + mt * 16 + ((l >> 4) << 3);
    v4f b0 = *(const v4f*)bp;
    v4f b1 = *(const v4f*)(bp + 4);
#pragma unroll
    for (int nt = 0; nt < 2; ++nt) {
#pragma unroll
      for (int r = 0; r < 4; ++r) { acc[mt][nt][r] = b0[r]; acc[mt][nt][r + 4] = b1[r]; }
    }
  }

  for (int k = 0; k < DIM; k += 32) {
    v16h a[4], b[2];
#pragma unroll
    for (int mt = 0; mt < 4; ++mt)
      a[mt] = load_A16x32(W + (size_t)(c0 + mt * 16) * DIM + k, DIM);
#pragma unroll
    for (int nt = 0; nt < 2; ++nt)
      b[nt] = load_B32x16(xh + (size_t)(t0 + nt * 16) * DIM + k, DIM);
#pragma unroll
    for (int mt = 0; mt < 4; ++mt)
#pragma unroll
      for (int nt = 0; nt < 2; ++nt)
        acc[mt][nt] = wmma_f16(a[mt], b[nt], acc[mt][nt]);
  }

#pragma unroll
  for (int mt = 0; mt < 4; ++mt)
#pragma unroll
    for (int nt = 0; nt < 2; ++nt) {
      int tg   = t0 + nt * 16 + (l & 15);
      int bs   = tg >> 11, tok = tg & 2047;
      int c    = c0 + mt * 16 + ((l >> 4) << 3);  // 8 consecutive channels
      int head = c >> 6,  hd  = c & 63;
      v8h o;
#pragma unroll
      for (int r = 0; r < 8; ++r) o[r] = (_Float16)acc[mt][nt][r];
      *(v8h*)(out + (((size_t)(bs * NHEAD + head) * NTOK + tok) * HD + hd)) = o;
    }
}

// ---------------------------------------------------------------------------
// V projection, direct:  D[m, o] -> stored as V^T[(bs*8+head)*64 + hd][tok]
// D layout: lane = channel, rows = token -> V^T store is contiguous v8h.
// Wave tile: 64 tokens x 32 channels. grid = (16384/64, 512/32)
// ---------------------------------------------------------------------------
__global__ void __launch_bounds__(32)
v_proj_kernel(const _Float16* __restrict__ xh, const _Float16* __restrict__ W,
              const float* __restrict__ bias, _Float16* __restrict__ vt) {
  const int l  = threadIdx.x & 31;
  const int m0 = blockIdx.x * 64;   // token group
  const int c0 = blockIdx.y * 32;   // channel group

  v8f acc[4][2];
#pragma unroll
  for (int nt = 0; nt < 2; ++nt) {
    float b = bias[c0 + nt * 16 + (l & 15)];
#pragma unroll
    for (int mt = 0; mt < 4; ++mt)
#pragma unroll
      for (int r = 0; r < 8; ++r) acc[mt][nt][r] = b;
  }

  for (int k = 0; k < DIM; k += 32) {
    v16h a[4], b[2];
#pragma unroll
    for (int mt = 0; mt < 4; ++mt)
      a[mt] = load_A16x32(xh + (size_t)(m0 + mt * 16) * DIM + k, DIM);
#pragma unroll
    for (int nt = 0; nt < 2; ++nt)
      b[nt] = load_B32x16(W + (size_t)(c0 + nt * 16) * DIM + k, DIM);
#pragma unroll
    for (int mt = 0; mt < 4; ++mt)
#pragma unroll
      for (int nt = 0; nt < 2; ++nt)
        acc[mt][nt] = wmma_f16(a[mt], b[nt], acc[mt][nt]);
  }

#pragma unroll
  for (int mt = 0; mt < 4; ++mt)
#pragma unroll
    for (int nt = 0; nt < 2; ++nt) {
      int c    = c0 + nt * 16 + (l & 15);
      int head = c >> 6, hd = c & 63;
      int mg   = m0 + mt * 16 + ((l >> 4) << 3);  // 8 consecutive tokens
      int bs   = mg >> 11, tok = mg & 2047;
      v8h o;
#pragma unroll
      for (int r = 0; r < 8; ++r) o[r] = (_Float16)acc[mt][nt][r];
      *(v8h*)(vt + (((size_t)(bs * NHEAD + head) * HD + hd) * NTOK + tok)) = o;
    }
}

// ---------------------------------------------------------------------------
// Flash attention, S^T / O^T formulation, 4 waves per block sharing K / V^T
// tiles staged in LDS via async global->LDS copies (double buffered).
//   S^T = K * Q^T          (A = K rows from LDS, B = Q^T columns from global)
//   O^T = V^T * P^T        (A = V^T rows from LDS, B = P^T built via lane^16)
// Tiles per 32-key stage: K 32x64 f16 (4KB, contiguous in global),
// V^T 64x32 f16 (4KB, 64 rows of 64B). Each wave issues 4 async-b128 per
// stage -> pipelined wait is s_wait_asynccnt 4.
// grid = (2048/128, 64), block = 128 (4 waves)
// ---------------------------------------------------------------------------
__global__ void __launch_bounds__(128)
attn_kernel(const _Float16* __restrict__ Qb, const _Float16* __restrict__ Kb,
            const _Float16* __restrict__ Vt, _Float16* __restrict__ Ob) {
  __shared__ _Float16 kbuf[2][32 * 64];
  __shared__ _Float16 vbuf[2][64 * 32];

  const int tid  = threadIdx.x;
  const int l    = tid & 31;
  const int wave = tid >> 5;
  const int bsh  = blockIdx.y;
  const int q0   = blockIdx.x * QBLK + wave * 32;
  const _Float16* Qh = Qb + (size_t)bsh * NTOK * HD;
  const _Float16* Kh = Kb + (size_t)bsh * NTOK * HD;
  const _Float16* Vh = Vt + (size_t)bsh * HD * NTOK;

  // stage issue: copy K[k0..k0+31][0..63] and V^T[0..63][k0..k0+31] into LDS
  auto issue_stage = [&](int buf, int k0) {
    const char* kg = (const char*)(Kh + (size_t)k0 * HD);
    unsigned int kl = (unsigned int)(uintptr_t)&kbuf[buf][0];
    async_b128(kl + tid * 16, kg + tid * 16);
    async_b128(kl + (tid + 128) * 16, kg + (tid + 128) * 16);

    const char* vg = (const char*)(Vh + k0);
    unsigned int vl = (unsigned int)(uintptr_t)&vbuf[buf][0];
    int c0 = tid, c1 = tid + 128;   // chunk = hd*4 + cc, row stride 4096B
    async_b128(vl + c0 * 16, vg + (size_t)(c0 >> 2) * 4096 + (c0 & 3) * 16);
    async_b128(vl + c1 * 16, vg + (size_t)(c1 >> 2) * 4096 + (c1 & 3) * 16);
  };

  issue_stage(0, 0);

  // Q^T B-fragments, cached for the whole pass: [qtile][hd-half]
  v16h qf[2][2];
#pragma unroll
  for (int qt = 0; qt < 2; ++qt)
#pragma unroll
    for (int kk = 0; kk < 2; ++kk)
      qf[qt][kk] = load_B32x16(Qh + (size_t)(q0 + qt * 16) * HD + kk * 32, HD);

  v8f acc[4][2];  // O^T tiles: [hd-tile][qtile], lane = query
#pragma unroll
  for (int h = 0; h < 4; ++h)
#pragma unroll
    for (int qt = 0; qt < 2; ++qt) acc[h][qt] = zero8();

  float mrow[2] = {-3.0e38f, -3.0e38f};
  float lrow[2] = {0.0f, 0.0f};
  const float sc = 0.0625f * 1.44269504088896f;  // (1/sqrt(64))*(1/2) * log2(e)
  const int half = l >> 4;

  for (int k0 = 0; k0 < NTOK; k0 += 32) {
    const int cur = (k0 >> 5) & 1;
    if (k0 + 32 < NTOK) {
      issue_stage(cur ^ 1, k0 + 32);
      asm volatile("s_wait_asynccnt 4" ::: "memory");   // stage `cur` complete
    } else {
      asm volatile("s_wait_asynccnt 0" ::: "memory");
    }
    __syncthreads();  // all waves' async data visible

    // ---- scores S^T tiles [keytile][qtile]; lane = query, rows = key
    v8f st[2][2];
#pragma unroll
    for (int kt = 0; kt < 2; ++kt) {
      v16h ka0 = load_A16x32(&kbuf[cur][kt * 16 * 64], 64);
      v16h ka1 = load_A16x32(&kbuf[cur][kt * 16 * 64 + 32], 64);
#pragma unroll
      for (int qt = 0; qt < 2; ++qt) {
        v8f s = zero8();
        s = wmma_f16(ka0, qf[qt][0], s);
        s = wmma_f16(ka1, qf[qt][1], s);
        st[kt][qt] = s;
      }
    }

    // ---- online softmax over keys (rows) + build P^T B-fragments
    v16h pb[2];
#pragma unroll
    for (int qt = 0; qt < 2; ++qt) {
      float mloc = -3.0e38f;
#pragma unroll
      for (int kt = 0; kt < 2; ++kt)
#pragma unroll
        for (int r = 0; r < 8; ++r) {
          float v = st[kt][qt][r] * sc;
          st[kt][qt][r] = v;
          mloc = fmaxf(mloc, v);
        }
      mloc = fmaxf(mloc, xor16f(mloc));          // combine lane halves
      float mnew = fmaxf(mrow[qt], mloc);
      float corr = exp2f(mrow[qt] - mnew);
      mrow[qt] = mnew;

      float psum = 0.0f;
#pragma unroll
      for (int kt = 0; kt < 2; ++kt)
#pragma unroll
        for (int r = 0; r < 8; ++r) {
          float p = exp2f(st[kt][qt][r] - mnew);
          st[kt][qt][r] = p;
          psum += p;
        }
      psum += xor16f(psum);
      lrow[qt] = lrow[qt] * corr + psum;

#pragma unroll
      for (int h = 0; h < 4; ++h)
#pragma unroll
        for (int r = 0; r < 8; ++r) acc[h][qt][r] *= corr;

      // P^T D-layout -> B-fragment: element e holds key e + 16*half.
      v16h p;
#pragma unroll
      for (int r = 0; r < 8; ++r) {
        float o0 = xor16f(st[0][qt][r]);   // partner's keys of tile0
        float o1 = xor16f(st[1][qt][r]);   // partner's keys of tile1
        float elo = half ? o1 : st[0][qt][r];   // k = r   + 16*half
        float ehi = half ? st[1][qt][r] : o0;   // k = r+8 + 16*half
        p[r]     = (_Float16)elo;
        p[r + 8] = (_Float16)ehi;
      }
      pb[qt] = p;
    }

    // ---- O^T += V^T * P^T
#pragma unroll
    for (int h = 0; h < 4; ++h) {
      v16h va = load_A16x32(&vbuf[cur][h * 16 * 32], 32);
#pragma unroll
      for (int qt = 0; qt < 2; ++qt)
        acc[h][qt] = wmma_f16(va, pb[qt], acc[h][qt]);
    }

    __syncthreads();  // all waves done with buf `cur` before it is refilled
  }

  // ---- epilogue: normalize and store row-major Ob[bs*2048+q][head*64+hd]
  const int bs = bsh >> 3, head = bsh & 7;
#pragma unroll
  for (int qt = 0; qt < 2; ++qt) {
    float inv = 1.0f / lrow[qt];
    int q = q0 + qt * 16 + (l & 15);
#pragma unroll
    for (int h = 0; h < 4; ++h) {
      v8h o;
#pragma unroll
      for (int r = 0; r < 8; ++r) o[r] = (_Float16)(acc[h][qt][r] * inv);
      *(v8h*)(Ob + ((size_t)(bs * NTOK + q) * DIM + head * HD + h * 16 +
                    ((l >> 4) << 3))) = o;
    }
  }
}

// ---------------------------------------------------------------------------
// Output projection + final [n,c] -> [c,n] transpose per (b,s).
// Direct GEMM: D lane = channel, rows = token -> storing to out[bs][c][tok]
// is contiguous (8 consecutive tokens per lane). f32 output.
// grid = (16384/64, 512/32)
// ---------------------------------------------------------------------------
__global__ void __launch_bounds__(32)
out_proj_kernel(const _Float16* __restrict__ Ob, const _Float16* __restrict__ W,
                const float* __restrict__ bias, float* __restrict__ out) {
  const int l  = threadIdx.x & 31;
  const int m0 = blockIdx.x * 64;
  const int c0 = blockIdx.y * 32;

  v8f acc[4][2];
#pragma unroll
  for (int nt = 0; nt < 2; ++nt) {
    float b = bias[c0 + nt * 16 + (l & 15)];
#pragma unroll
    for (int mt = 0; mt < 4; ++mt)
#pragma unroll
      for (int r = 0; r < 8; ++r) acc[mt][nt][r] = b;
  }

  for (int k = 0; k < DIM; k += 32) {
    v16h a[4], b[2];
#pragma unroll
    for (int mt = 0; mt < 4; ++mt)
      a[mt] = load_A16x32(Ob + (size_t)(m0 + mt * 16) * DIM + k, DIM);
#pragma unroll
    for (int nt = 0; nt < 2; ++nt)
      b[nt] = load_B32x16(W + (size_t)(c0 + nt * 16) * DIM + k, DIM);
#pragma unroll
    for (int mt = 0; mt < 4; ++mt)
#pragma unroll
      for (int nt = 0; nt < 2; ++nt)
        acc[mt][nt] = wmma_f16(a[mt], b[nt], acc[mt][nt]);
  }

#pragma unroll
  for (int mt = 0; mt < 4; ++mt)
#pragma unroll
    for (int nt = 0; nt < 2; ++nt) {
      int c  = c0 + nt * 16 + (l & 15);
      int mg = m0 + mt * 16 + ((l >> 4) << 3);
      int bs = mg >> 11, tok = mg & 2047;
      float* p = out + ((size_t)(bs * DIM + c) * NTOK + tok);
      v4f lo, hi;
#pragma unroll
      for (int r = 0; r < 4; ++r) { lo[r] = acc[mt][nt][r]; hi[r] = acc[mt][nt][r + 4]; }
      *(v4f*)p = lo;
      *(v4f*)(p + 4) = hi;
    }
}

// ---------------------------------------------------------------------------
extern "C" void kernel_launch(void* const* d_in, const int* in_sizes, int n_in,
                              void* d_out, int out_size, void* d_ws,
                              size_t ws_size, hipStream_t stream) {
  const float* x  = (const float*)d_in[0];
  const float* Wq = (const float*)d_in[1];
  const float* bq = (const float*)d_in[2];
  const float* Wk = (const float*)d_in[3];
  const float* bk = (const float*)d_in[4];
  const float* Wv = (const float*)d_in[5];
  const float* bv = (const float*)d_in[6];
  const float* Wo = (const float*)d_in[7];
  const float* bo = (const float*)d_in[8];
  float* out = (float*)d_out;

  char* ws = (char*)d_ws;
  size_t off = 0;
  _Float16* xh  = (_Float16*)(ws + off); off += (size_t)MROWS * DIM * 2;
  _Float16* Wqh = (_Float16*)(ws + off); off += (size_t)DIM * DIM * 2;
  _Float16* Wkh = (_Float16*)(ws + off); off += (size_t)DIM * DIM * 2;
  _Float16* Wvh = (_Float16*)(ws + off); off += (size_t)DIM * DIM * 2;
  _Float16* Woh = (_Float16*)(ws + off); off += (size_t)DIM * DIM * 2;
  _Float16* Qb  = (_Float16*)(ws + off); off += (size_t)MROWS * DIM * 2;
  _Float16* Kb  = (_Float16*)(ws + off); off += (size_t)MROWS * DIM * 2;
  _Float16* Vt  = (_Float16*)(ws + off); off += (size_t)MROWS * DIM * 2;
  _Float16* Ob  = (_Float16*)(ws + off); off += (size_t)MROWS * DIM * 2;

  const int xN4 = (MROWS * DIM) / 4;     // 2,097,152
  const int wN4 = (DIM * DIM) / 4;       // 65,536
  cvt_f32_f16<<<(xN4 + 255) / 256, 256, 0, stream>>>(x, xh, xN4);
  cvt_f32_f16<<<(wN4 + 255) / 256, 256, 0, stream>>>(Wq, Wqh, wN4);
  cvt_f32_f16<<<(wN4 + 255) / 256, 256, 0, stream>>>(Wk, Wkh, wN4);
  cvt_f32_f16<<<(wN4 + 255) / 256, 256, 0, stream>>>(Wv, Wvh, wN4);
  cvt_f32_f16<<<(wN4 + 255) / 256, 256, 0, stream>>>(Wo, Woh, wN4);

  qk_proj_kernel<<<dim3(MROWS / 32, DIM / 64), 32, 0, stream>>>(xh, Wqh, bq, Qb);
  qk_proj_kernel<<<dim3(MROWS / 32, DIM / 64), 32, 0, stream>>>(xh, Wkh, bk, Kb);
  v_proj_kernel <<<dim3(MROWS / 64, DIM / 32), 32, 0, stream>>>(xh, Wvh, bv, Vt);

  attn_kernel<<<dim3(NTOK / QBLK, BSZ * NHEAD), 128, 0, stream>>>(Qb, Kb, Vt, Ob);

  out_proj_kernel<<<dim3(MROWS / 64, DIM / 32), 32, 0, stream>>>(Ob, Woh, bo, out);
}
